// MultiHeadAttention_7206955123107
// MI455X (gfx1250) — compile-verified, matrix operand order checked by reading, not run on hardware
//
#include <hip/hip_runtime.h>
#include <hip/hip_bf16.h>
#include <math.h>

typedef __attribute__((ext_vector_type(16))) _Float16 v16h;
typedef __attribute__((ext_vector_type(8)))  float    v8f;
typedef __attribute__((ext_vector_type(4)))  unsigned int u32x4;
typedef __attribute__((ext_vector_type(8)))  int      i32x8;
typedef __attribute__((ext_vector_type(4)))  int      i32x4;

#define S_LEN 4096
#define DM    1024
#define NH    16
#define DKH   64   // head dim

// ---------------------------------------------------------------------------
// Fragment loader (global or LDS) with contiguous k order (k = 16*(lane/16) + s), same for A and B.
// A (16x32 f16): lanes 0-15 -> row (lane&15), K 0..15 ; lanes 16-31 -> K 16..31
// B (32x16 f16): identical addressing against B^T stored [N, K] row-major.
// ---------------------------------------------------------------------------
__device__ __forceinline__ v16h load_frag(const _Float16* base, int ld) {
  int l = threadIdx.x & 31;
  return *(const v16h*)(base + (long)(l & 15) * ld + ((l >> 4) << 4));
}

__device__ __forceinline__ v8f wmma_f16(v16h a, v16h b, v8f c) {
  return __builtin_amdgcn_wmma_f32_16x16x32_f16(false, a, false, b, (short)0, c,
                                                false, false);
}

// ---------------------------------------------------------------------------
// Tensor Data Mover helpers. D# group1 (dims/strides, constant per tensor) is
// built once and hoisted; per-issue we only pack the new global address.
// ---------------------------------------------------------------------------
__device__ __forceinline__ i32x8 tdm_g1_2d(unsigned int tensor_d0,
                                           unsigned int tensor_d1,
                                           unsigned int tile_d0,
                                           unsigned int tile_d1,
                                           unsigned int stride0) {
  i32x8 g1 = { (int)0x00010000u,                                  // data_size=2B
               (int)((tensor_d0 & 0xFFFFu) << 16),
               (int)((tensor_d0 >> 16) | ((tensor_d1 & 0xFFFFu) << 16)),
               (int)((tensor_d1 >> 16) | (tile_d0 << 16)),
               (int)tile_d1,                                      // tile_dim2=0
               (int)stride0,
               0, 0 };
  return g1;
}

__device__ __forceinline__ void tdm_issue(unsigned int lds_addr,
                                          const void* gptr, i32x8 g1) {
  unsigned long long ga = (unsigned long long)gptr;
  u32x4 g0 = { 1u, lds_addr, (unsigned int)ga,
               (unsigned int)((ga >> 32) & 0x1FFFFFFu) | 0x80000000u };
  i32x4 z4 = {};
  i32x8 z8 = {};
  __builtin_amdgcn_tensor_load_to_lds(g0, g1, z4, z4, z8, 0);
}

// ---------------------------------------------------------------------------
// Elementwise converts
// ---------------------------------------------------------------------------
__global__ void cvt_f32_f16(const float* __restrict__ src,
                            _Float16* __restrict__ dst, int n) {
  int i = blockIdx.x * blockDim.x + threadIdx.x;
  if (i < n) dst[i] = (_Float16)src[i];
}

// W: [NH, DM, DKH] f32  ->  Wt: [NH, DKH, DM] f16  (B^T per head)
__global__ void transpose_w(const float* __restrict__ W,
                            _Float16* __restrict__ Wt) {
  int i = blockIdx.x * blockDim.x + threadIdx.x;
  if (i >= NH * DM * DKH) return;
  int dk = i % DKH;
  int d  = (i / DKH) % DM;
  int h  = i / (DKH * DM);
  Wt[((long)h * DKH + dk) * DM + d] = (_Float16)W[i];
}

// ---------------------------------------------------------------------------
// QKV projection GEMM: dst[h] = (A[S,DM] x W[h][DM,DKH] + b[h]) * scale
// 8 waves, tile 256(M) x 64(N); each wave 32x64; K-loop unrolled x2 with
// explicit ping-pong stage registers (no rotation copies).
// ---------------------------------------------------------------------------
__global__ __launch_bounds__(256)
void proj_gemm(const _Float16* __restrict__ A,
               const _Float16* __restrict__ Bt,
               const float* __restrict__ bias,
               _Float16* __restrict__ dst,
               float scale, int transposed) {
  const int wave = threadIdx.x >> 5;
  const int lane = threadIdx.x & 31;
  const int h    = blockIdx.x;
  const int m0   = blockIdx.y * 256 + wave * 32;

  v8f acc[2][4];
#pragma unroll
  for (int r = 0; r < 2; ++r)
#pragma unroll
    for (int t = 0; t < 4; ++t) acc[r][t] = (v8f){};

  const _Float16* A0 = A + (long)m0 * DM;
  const _Float16* A1 = A0 + 16 * DM;
  const _Float16* Bb = Bt + (long)h * DKH * DM;

  v16h a0[2], a1[2], b[2][4];
  a0[0] = load_frag(A0, DM);
  a1[0] = load_frag(A1, DM);
#pragma unroll
  for (int t = 0; t < 4; ++t) b[0][t] = load_frag(Bb + (long)(t * 16) * DM, DM);

  for (int k0 = 0; k0 < DM; k0 += 64) {
    // prefetch stage 1 (k0+32 always < DM)
    a0[1] = load_frag(A0 + k0 + 32, DM);
    a1[1] = load_frag(A1 + k0 + 32, DM);
#pragma unroll
    for (int t = 0; t < 4; ++t)
      b[1][t] = load_frag(Bb + (long)(t * 16) * DM + k0 + 32, DM);

#pragma unroll
    for (int t = 0; t < 4; ++t) {
      acc[0][t] = wmma_f16(a0[0], b[0][t], acc[0][t]);
      acc[1][t] = wmma_f16(a1[0], b[0][t], acc[1][t]);
    }

    // prefetch stage 0 for next pair (clamped dummy on the last pair)
    const int kn = (k0 + 64 < DM) ? (k0 + 64) : 0;
    a0[0] = load_frag(A0 + kn, DM);
    a1[0] = load_frag(A1 + kn, DM);
#pragma unroll
    for (int t = 0; t < 4; ++t)
      b[0][t] = load_frag(Bb + (long)(t * 16) * DM + kn, DM);

#pragma unroll
    for (int t = 0; t < 4; ++t) {
      acc[0][t] = wmma_f16(a0[1], b[1][t], acc[0][t]);
      acc[1][t] = wmma_f16(a1[1], b[1][t], acc[1][t]);
    }
  }

  const int rbase = (lane >> 4) << 3;
#pragma unroll
  for (int r = 0; r < 2; ++r)
#pragma unroll
    for (int t = 0; t < 4; ++t) {
      int dk = t * 16 + (lane & 15);
      float bv = bias[h * DKH + dk];
#pragma unroll
      for (int i = 0; i < 8; ++i) {
        int m = m0 + r * 16 + rbase + i;
        float val = (acc[r][t][i] + bv) * scale;
        if (!transposed)
          dst[((long)h * S_LEN + m) * DKH + dk] = (_Float16)val;
        else
          dst[((long)h * DKH + dk) * S_LEN + m] = (_Float16)val;
      }
    }
}

// ---------------------------------------------------------------------------
// Flash attention: one (head, 128-query-row block) per 256-thread block.
// K/V 64x64 tiles DMA'd into double-buffered LDS by the TDM (wave 0), synced
// with s_wait_tensorcnt + barrier; all 8 waves read fragments from LDS.
// q is pre-scaled by 1/sqrt(dk) at projection time.
// ---------------------------------------------------------------------------
__global__ __launch_bounds__(256)
void flash_attn(const _Float16* __restrict__ q,
                const _Float16* __restrict__ k,
                const _Float16* __restrict__ vT,
                _Float16* __restrict__ x) {
  __shared__ __align__(32) _Float16 kt[2][64][64];     // [key][dk]
  __shared__ __align__(32) _Float16 vt[2][64][64];     // [dk][key]
  __shared__ __align__(32) _Float16 ldsP[8][16][64];   // per-wave P tile

  const int wave = threadIdx.x >> 5;
  const int lane = threadIdx.x & 31;
  const int h    = blockIdx.y;
  const int m0   = blockIdx.x * 128 + wave * 16;

  const _Float16* qb = q  + ((long)h * S_LEN + m0) * DKH;
  const _Float16* kb = k  + (long)h * S_LEN * DKH;
  const _Float16* vb = vT + (long)h * DKH * S_LEN;

  const v16h aq0 = load_frag(qb, DKH);
  const v16h aq1 = load_frag(qb + 32, DKH);

  v8f o[4];
#pragma unroll
  for (int t = 0; t < 4; ++t) o[t] = (v8f){};

  float mrow[8], lrow[8];
#pragma unroll
  for (int i = 0; i < 8; ++i) { mrow[i] = -1e30f; lrow[i] = 0.0f; }

  const int rbase = (lane >> 4) << 3;

  // constant D# group1 descriptors, hoisted out of the loop
  const i32x8 g1k = tdm_g1_2d(DKH, S_LEN, 64, 64, DKH);
  const i32x8 g1v = tdm_g1_2d(S_LEN, DKH, 64, 64, S_LEN);
  const unsigned int ldsK[2] = { (unsigned int)(unsigned long long)&kt[0][0][0],
                                 (unsigned int)(unsigned long long)&kt[1][0][0] };
  const unsigned int ldsV[2] = { (unsigned int)(unsigned long long)&vt[0][0][0],
                                 (unsigned int)(unsigned long long)&vt[1][0][0] };

  if (wave == 0) {                              // prime buffer 0
    tdm_issue(ldsK[0], kb, g1k);
    tdm_issue(ldsV[0], vb, g1v);
  }

  for (int j0 = 0; j0 < S_LEN; j0 += 64) {
    const int cur = (j0 >> 6) & 1;
    const int nxt = cur ^ 1;
    if (wave == 0) {
      if (j0 + 64 < S_LEN) {
        tdm_issue(ldsK[nxt], kb + (long)(j0 + 64) * DKH, g1k);
        tdm_issue(ldsV[nxt], vb + (j0 + 64), g1v);
        __builtin_amdgcn_s_wait_tensorcnt(2);   // current pair complete
      } else {
        __builtin_amdgcn_s_wait_tensorcnt(0);
      }
    }
    __syncthreads();                            // tiles visible to all waves

    // ---- scores (q pre-scaled) ------------------------------------------
    v8f s[4];
#pragma unroll
    for (int t = 0; t < 4; ++t) {
      v16h b0 = load_frag(&kt[cur][t * 16][0], 64);
      v16h b1 = load_frag(&kt[cur][t * 16][32], 64);
      v8f c = (v8f){};
      c = wmma_f16(aq0, b0, c);
      c = wmma_f16(aq1, b1, c);
      s[t] = c;
    }

    // ---- online softmax (rows striped over 16-lane halves) ---------------
#pragma unroll
    for (int i = 0; i < 8; ++i) {
      float mx = fmaxf(fmaxf(s[0][i], s[1][i]), fmaxf(s[2][i], s[3][i]));
      mx = fmaxf(mx, __shfl_xor(mx, 1, 32));
      mx = fmaxf(mx, __shfl_xor(mx, 2, 32));
      mx = fmaxf(mx, __shfl_xor(mx, 4, 32));
      mx = fmaxf(mx, __shfl_xor(mx, 8, 32));

      float mnew = fmaxf(mrow[i], mx);
      float corr = __expf(mrow[i] - mnew);
      float rsum = 0.0f;
#pragma unroll
      for (int t = 0; t < 4; ++t) {
        float p = __expf(s[t][i] - mnew);
        s[t][i] = p;
        rsum += p;
      }
      rsum += __shfl_xor(rsum, 1, 32);
      rsum += __shfl_xor(rsum, 2, 32);
      rsum += __shfl_xor(rsum, 4, 32);
      rsum += __shfl_xor(rsum, 8, 32);

      lrow[i] = lrow[i] * corr + rsum;
      mrow[i] = mnew;
#pragma unroll
      for (int t = 0; t < 4; ++t) o[t][i] *= corr;
    }

    // ---- re-layout P: C layout -> A fragments via per-wave LDS -----------
#pragma unroll
    for (int t = 0; t < 4; ++t)
#pragma unroll
      for (int i = 0; i < 8; ++i)
        ldsP[wave][rbase + i][t * 16 + (lane & 15)] = (_Float16)s[t][i];

    asm volatile("s_wait_dscnt 0" ::: "memory");

    v16h pa0 = load_frag(&ldsP[wave][0][0], 64);
    v16h pa1 = load_frag(&ldsP[wave][0][32], 64);

    // ---- O += P x V ------------------------------------------------------
#pragma unroll
    for (int t = 0; t < 4; ++t) {
      v16h b0 = load_frag(&vt[cur][t * 16][0], 64);
      v16h b1 = load_frag(&vt[cur][t * 16][32], 64);
      o[t] = wmma_f16(pa0, b0, o[t]);
      o[t] = wmma_f16(pa1, b1, o[t]);
    }
    __syncthreads();   // all reads of buf[cur] done before TDM reuses it
  }

  // ---- epilogue ----------------------------------------------------------
#pragma unroll
  for (int t = 0; t < 4; ++t) {
    int col = h * DKH + t * 16 + (lane & 15);
#pragma unroll
    for (int i = 0; i < 8; ++i) {
      int m = m0 + rbase + i;
      x[(long)m * DM + col] = (_Float16)(o[t][i] / lrow[i]);
    }
  }
}

// ---------------------------------------------------------------------------
// Output projection: out[S,DM] = x[S,DM] @ Wo^T + bo, f32 output.
// Same ping-pong double-buffered structure.
// ---------------------------------------------------------------------------
__global__ __launch_bounds__(256)
void out_gemm(const _Float16* __restrict__ A,
              const _Float16* __restrict__ Bt,
              const float* __restrict__ bias,
              float* __restrict__ dst) {
  const int wave = threadIdx.x >> 5;
  const int lane = threadIdx.x & 31;
  const int m0   = blockIdx.y * 256 + wave * 32;
  const int n0   = blockIdx.x * 64;

  v8f acc[2][4];
#pragma unroll
  for (int r = 0; r < 2; ++r)
#pragma unroll
    for (int t = 0; t < 4; ++t) acc[r][t] = (v8f){};

  const _Float16* A0 = A + (long)m0 * DM;
  const _Float16* A1 = A0 + 16 * DM;
  const _Float16* Bb = Bt + (long)n0 * DM;

  v16h a0[2], a1[2], b[2][4];
  a0[0] = load_frag(A0, DM);
  a1[0] = load_frag(A1, DM);
#pragma unroll
  for (int t = 0; t < 4; ++t) b[0][t] = load_frag(Bb + (long)(t * 16) * DM, DM);

  for (int k0 = 0; k0 < DM; k0 += 64) {
    a0[1] = load_frag(A0 + k0 + 32, DM);
    a1[1] = load_frag(A1 + k0 + 32, DM);
#pragma unroll
    for (int t = 0; t < 4; ++t)
      b[1][t] = load_frag(Bb + (long)(t * 16) * DM + k0 + 32, DM);

#pragma unroll
    for (int t = 0; t < 4; ++t) {
      acc[0][t] = wmma_f16(a0[0], b[0][t], acc[0][t]);
      acc[1][t] = wmma_f16(a1[0], b[0][t], acc[1][t]);
    }

    const int kn = (k0 + 64 < DM) ? (k0 + 64) : 0;
    a0[0] = load_frag(A0 + kn, DM);
    a1[0] = load_frag(A1 + kn, DM);
#pragma unroll
    for (int t = 0; t < 4; ++t)
      b[0][t] = load_frag(Bb + (long)(t * 16) * DM + kn, DM);

#pragma unroll
    for (int t = 0; t < 4; ++t) {
      acc[0][t] = wmma_f16(a0[1], b[1][t], acc[0][t]);
      acc[1][t] = wmma_f16(a1[1], b[1][t], acc[1][t]);
    }
  }

  const int rbase = (lane >> 4) << 3;
#pragma unroll
  for (int r = 0; r < 2; ++r)
#pragma unroll
    for (int t = 0; t < 4; ++t) {
      int n = n0 + t * 16 + (lane & 15);
      float bv = bias[n];
#pragma unroll
      for (int i = 0; i < 8; ++i) {
        int m = m0 + r * 16 + rbase + i;
        dst[(long)m * DM + n] = acc[r][t][i] + bv;
      }
    }
}

// ---------------------------------------------------------------------------
extern "C" void kernel_launch(void* const* d_in, const int* in_sizes, int n_in,
                              void* d_out, int out_size, void* d_ws, size_t ws_size,
                              hipStream_t stream) {
  const float* Q  = (const float*)d_in[0];
  const float* K  = (const float*)d_in[1];
  const float* V  = (const float*)d_in[2];
  const float* Wq = (const float*)d_in[3];
  const float* bq = (const float*)d_in[4];
  const float* Wk = (const float*)d_in[5];
  const float* bk = (const float*)d_in[6];
  const float* Wv = (const float*)d_in[7];
  const float* bv = (const float*)d_in[8];
  const float* Wo = (const float*)d_in[9];
  const float* bo = (const float*)d_in[10];
  float* out = (float*)d_out;

  char* base = (char*)d_ws;
  size_t off = 0;
  auto alloc = [&](size_t elems) -> _Float16* {
    _Float16* p = (_Float16*)(base + off);
    off += (elems * sizeof(_Float16) + 255) & ~(size_t)255;
    return p;
  };

  _Float16* Qh  = alloc((size_t)S_LEN * DM);
  _Float16* Kh  = alloc((size_t)S_LEN * DM);
  _Float16* Vh  = alloc((size_t)S_LEN * DM);
  _Float16* Wtq = alloc((size_t)NH * DKH * DM);
  _Float16* Wtk = alloc((size_t)NH * DKH * DM);
  _Float16* Wtv = alloc((size_t)NH * DKH * DM);
  _Float16* Woh = alloc((size_t)DM * DM);
  _Float16* qh  = alloc((size_t)NH * S_LEN * DKH);  // [H,S,DK]
  _Float16* kh  = alloc((size_t)NH * S_LEN * DKH);  // [H,S,DK]
  _Float16* vTh = alloc((size_t)NH * DKH * S_LEN);  // [H,DK,S]
  _Float16* xh  = alloc((size_t)S_LEN * DM);

  const int nQKV = S_LEN * DM;
  const int nW   = NH * DM * DKH;
  const int nWo  = DM * DM;

  cvt_f32_f16<<<(nQKV + 255) / 256, 256, 0, stream>>>(Q, Qh, nQKV);
  cvt_f32_f16<<<(nQKV + 255) / 256, 256, 0, stream>>>(K, Kh, nQKV);
  cvt_f32_f16<<<(nQKV + 255) / 256, 256, 0, stream>>>(V, Vh, nQKV);
  cvt_f32_f16<<<(nWo + 255) / 256, 256, 0, stream>>>(Wo, Woh, nWo);
  transpose_w<<<(nW + 255) / 256, 256, 0, stream>>>(Wq, Wtq);
  transpose_w<<<(nW + 255) / 256, 256, 0, stream>>>(Wk, Wtk);
  transpose_w<<<(nW + 255) / 256, 256, 0, stream>>>(Wv, Wtv);

  dim3 gProj(NH, S_LEN / 256);                 // (16, 16)
  proj_gemm<<<gProj, 256, 0, stream>>>(Qh, Wtq, bq, qh, 0.125f, 0);  // q * 1/sqrt(dk)
  proj_gemm<<<gProj, 256, 0, stream>>>(Kh, Wtk, bk, kh, 1.0f, 0);
  proj_gemm<<<gProj, 256, 0, stream>>>(Vh, Wtv, bv, vTh, 1.0f, 1);

  dim3 gAttn(S_LEN / 128, NH);                 // (32, 16)
  flash_attn<<<gAttn, 256, 0, stream>>>(qh, kh, vTh, xh);

  dim3 gOut(DM / 64, S_LEN / 256);             // (16, 16)
  out_gemm<<<gOut, 256, 0, stream>>>(xh, Woh, bo, out);
}
